// CNN_geotransform_21165598834700
// MI455X (gfx1250) — matrix operand order repaired and checked by reading, not compiled, not run on hardware
//
#include <hip/hip_runtime.h>
#include <hip/hip_bf16.h>

typedef __attribute__((ext_vector_type(16))) _Float16 v16h;
typedef __attribute__((ext_vector_type(8)))  _Float16 v8h;
typedef __attribute__((ext_vector_type(8)))  float    v8f;

#define WAVES_PER_BLOCK 4
#define TB (WAVES_PER_BLOCK * 32)

__device__ __forceinline__ v8h ldv8h(const _Float16* p) {
  return *reinterpret_cast<const v8h*>(p);
}

__device__ __forceinline__ v16h make_frag(v8h lo, v8h hi) {
  v16h f;
#pragma unroll
  for (int i = 0; i < 8; ++i) { f[i] = lo[i]; f[i + 8] = hi[i]; }
  return f;
}

__device__ __forceinline__ v8f wmma_f16(v16h a, v16h b, v8f c) {
  return __builtin_amdgcn_wmma_f32_16x16x32_f16(false, a, false, b, (short)0,
                                                c, false, false);
}

// ---------------------------------------------------------------------------
// Weight staging: HWIO f32 -> transposed f16 [Cout, K], K = taps*CinP with
// zero padding CinS -> CinP. Columns of GEMM-B become contiguous rows so the
// WMMA B fragment is two b128 loads per lane.
// ---------------------------------------------------------------------------
__global__ void stage_wT_kernel(const float* __restrict__ src,
                                _Float16* __restrict__ dst,
                                int taps, int CinS, int CinP, int Cout) {
  const int K = taps * CinP;
  const int total = Cout * K;
  int idx = blockIdx.x * blockDim.x + threadIdx.x;
  if (idx >= total) return;
  const int n  = idx / K;
  const int kk = idx - n * K;
  const int t  = kk / CinP;
  const int ci = kk - t * CinP;
  const float v = (ci < CinS) ? src[((size_t)t * CinS + ci) * Cout + n] : 0.f;
  dst[idx] = (_Float16)v;
}

// ---------------------------------------------------------------------------
// conv1: 3x3x3 -> 64, stride 2 SAME (pad_before = 0), relu, f16 out. K = 27
// is too small for WMMA; direct VALU conv, bandwidth-trivial.
// ---------------------------------------------------------------------------
__global__ void conv1_direct_kernel(const float* __restrict__ img,
                                    const float* __restrict__ w,
                                    const float* __restrict__ bias,
                                    _Float16* __restrict__ out) {
  int idx = blockIdx.x * blockDim.x + threadIdx.x;
  const int total = 16 * 120 * 120 * 64;
  if (idx >= total) return;
  int co = idx & 63;
  int t  = idx >> 6;
  int ox = t % 120; t /= 120;
  int oy = t % 120;
  int b  = t / 120;
  float acc = bias[co];
  for (int ky = 0; ky < 3; ++ky) {
    int iy = oy * 2 + ky;
    if (iy >= 240) continue;
    for (int kx = 0; kx < 3; ++kx) {
      int ix = ox * 2 + kx;
      if (ix >= 240) continue;
      const float* ip = img + (((size_t)b * 240 + iy) * 240 + ix) * 3;
      const float* wp = w + ((ky * 3 + kx) * 3) * 64 + co;
      acc += ip[0] * wp[0] + ip[1] * wp[64] + ip[2] * wp[128];
    }
  }
  out[idx] = (_Float16)fmaxf(acc, 0.f);
}

// ---------------------------------------------------------------------------
// Implicit-GEMM conv via WMMA f32 = f16 x f16.
//   A (M x K): M = output pixels, K = tap*Cin + ci (Cin % 32 == 0)
//   B (K x N): weights pre-transposed [Cout, K] so columns are contiguous
// Each wave: one 16(M) x 64(N) strip = 4 accumulators, A fragment reused x4.
// K loop: taps outer (hoists all addressing), ci inner in steps of 32.
// Epilogue: +bias, relu, optional inference BN; f16 store.
// ---------------------------------------------------------------------------
__global__ __launch_bounds__(TB)
void conv_wmma_kernel(const _Float16* __restrict__ in,
                      const _Float16* __restrict__ wT,
                      const float* __restrict__ bias,
                      const float* __restrict__ bn_g,
                      const float* __restrict__ bn_b,
                      const float* __restrict__ bn_m,
                      const float* __restrict__ bn_v,
                      _Float16* __restrict__ out,
                      int Hin, int Win, int Cin,
                      int Hout, int Wout, int Cout,
                      int KH, int stride, int use_bn) {
  const int lane = threadIdx.x & 31;
  const int wv   = threadIdx.x >> 5;
  const int half = lane >> 4;
  const int l16  = lane & 15;
  const int b    = blockIdx.z;
  const int M    = Hout * Wout;
  const size_t K = (size_t)KH * KH * Cin;

  const int mbase = (blockIdx.x * WAVES_PER_BLOCK + wv) * 16;
  if (mbase >= M) return;                      // wave-uniform exit
  const int nbase = blockIdx.y * 64;           // Cout % 64 == 0 for all convs

  const _Float16* inb  = in  + (size_t)b * Hin * Win * Cin;
  _Float16*       outb = out + (size_t)b * M * Cout;

  int mrow = mbase + l16;
  mrow = (mrow < M) ? mrow : (M - 1);          // clamp: extra rows discarded
  const int oy = mrow / Wout;
  const int ox = mrow - oy * Wout;

  // B fragment base: lane holds column n = nbase+l16 (+t*16), K runs of 8
  // at half*8 and 16+half*8 within each 32-wide K step.
  const _Float16* bcol = wT + (size_t)(nbase + l16) * K + half * 8;

  v8f acc[4] = {};
  for (int ky = 0; ky < KH; ++ky) {
    const int iy = oy * stride + ky;
    for (int kx = 0; kx < KH; ++kx) {
      const int ix = ox * stride + kx;
      const bool valid = (iy < Hin) && (ix < Win);   // SAME-pad OOB -> zeros
      const _Float16* ap = inb + ((size_t)iy * Win + ix) * Cin + half * 8;
      const _Float16* wp = bcol + (size_t)(ky * KH + kx) * Cin;
      for (int ci = 0; ci < Cin; ci += 32) {
        const v8h z = {};
        const v8h a0 = valid ? ldv8h(ap + ci)      : z;
        const v8h a1 = valid ? ldv8h(ap + ci + 16) : z;
        const v16h afrag = make_frag(a0, a1);
        __builtin_prefetch(ap + ci + 256, 0, 1);
#pragma unroll
        for (int t = 0; t < 4; ++t) {
          const _Float16* w0 = wp + (size_t)t * 16 * K + ci;
          acc[t] = wmma_f16(afrag, make_frag(ldv8h(w0), ldv8h(w0 + 16)), acc[t]);
        }
      }
    }
  }

  // C layout: VGPR r, lanes 0-15 -> M=r, lanes 16-31 -> M=r+8; N = lane&15.
#pragma unroll
  for (int t = 0; t < 4; ++t) {
    const int n = nbase + t * 16 + l16;
    const float sc = use_bn ? (bn_g[n] * rsqrtf(bn_v[n] + 1e-3f)) : 1.f;
#pragma unroll
    for (int r = 0; r < 8; ++r) {
      const int m = mbase + r + half * 8;
      if (m < M) {
        float y = acc[t][r] + bias[n];
        y = fmaxf(y, 0.f);
        if (use_bn) y = (y - bn_m[n]) * sc + bn_b[n];
        outb[(size_t)m * Cout + n] = (_Float16)y;
      }
    }
  }
}

// ---------------------------------------------------------------------------
// Batched correlation: corr[b,m,n] = sum_c fA[b,m,c] * fB[b,n,c].
// Both A rows and (implicit) B columns are contiguous in c -> pure b128
// fragment loads. Wave: 16 x 64 strip; output row stride PS (channel-padded).
// ---------------------------------------------------------------------------
__global__ __launch_bounds__(TB)
void corr_wmma_kernel(const _Float16* __restrict__ fA,
                      const _Float16* __restrict__ fB,
                      _Float16* __restrict__ corr,
                      int P, int C, int PS) {
  const int lane = threadIdx.x & 31;
  const int wv   = threadIdx.x >> 5;
  const int half = lane >> 4;
  const int l16  = lane & 15;
  const int b    = blockIdx.z;

  const int mbase = (blockIdx.x * WAVES_PER_BLOCK + wv) * 16;
  if (mbase >= P) return;
  const int nbase = blockIdx.y * 64;

  const _Float16* Ab = fA + (size_t)b * P * C;
  const _Float16* Bb = fB + (size_t)b * P * C;

  int mrow = mbase + l16;
  mrow = (mrow < P) ? mrow : (P - 1);
  const _Float16* Ar = Ab + (size_t)mrow * C + half * 8;

  const _Float16* Br[4];
#pragma unroll
  for (int t = 0; t < 4; ++t) {
    int nc = nbase + t * 16 + l16;
    nc = (nc < P) ? nc : (P - 1);
    Br[t] = Bb + (size_t)nc * C + half * 8;
  }

  v8f acc[4] = {};
  for (int k0 = 0; k0 < C; k0 += 32) {
    const v16h afrag = make_frag(ldv8h(Ar + k0), ldv8h(Ar + k0 + 16));
#pragma unroll
    for (int t = 0; t < 4; ++t)
      acc[t] = wmma_f16(afrag, make_frag(ldv8h(Br[t] + k0), ldv8h(Br[t] + k0 + 16)),
                        acc[t]);
  }

#pragma unroll
  for (int t = 0; t < 4; ++t) {
    const int n = nbase + t * 16 + l16;
#pragma unroll
    for (int r = 0; r < 8; ++r) {
      const int m = mbase + r + half * 8;
      if (m < P && n < P)
        corr[((size_t)b * P + m) * PS + n] = (_Float16)acc[t][r];
    }
  }
}

// ---------------------------------------------------------------------------
// Zero the channel padding columns [len, stride) of each row.
// ---------------------------------------------------------------------------
__global__ void pad_zero_kernel(_Float16* __restrict__ x, int rows, int len,
                                int stride) {
  const int pad = stride - len;
  const int total = rows * pad;
  int idx = blockIdx.x * blockDim.x + threadIdx.x;
  if (idx >= total) return;
  const int r = idx / pad;
  const int c = idx - r * pad;
  x[(size_t)r * stride + len + c] = (_Float16)0.f;
}

// ---------------------------------------------------------------------------
// In-place row L2-normalize (optional relu first); rows have given stride.
// ---------------------------------------------------------------------------
__global__ void rownorm_kernel(_Float16* __restrict__ x, int len, int stride,
                               int do_relu, float eps) {
  __shared__ float red[256];
  _Float16* p = x + (size_t)blockIdx.x * stride;
  float s = 0.f;
  for (int i = threadIdx.x; i < len; i += blockDim.x) {
    float v = (float)p[i];
    if (do_relu) v = fmaxf(v, 0.f);
    s += v * v;
  }
  red[threadIdx.x] = s;
  __syncthreads();
  for (int off = 128; off > 0; off >>= 1) {
    if ((int)threadIdx.x < off) red[threadIdx.x] += red[threadIdx.x + off];
    __syncthreads();
  }
  const float scale = rsqrtf(red[0] + eps);
  for (int i = threadIdx.x; i < len; i += blockDim.x) {
    float v = (float)p[i];
    if (do_relu) v = fmaxf(v, 0.f);
    p[i] = (_Float16)(v * scale);
  }
}

// ---------------------------------------------------------------------------
// Dense head: out[b,j] = sum_i x[b,i]*dw[i,j] + db[j]   (16 x 25600 x 18)
// ---------------------------------------------------------------------------
__global__ void dense_kernel(const _Float16* __restrict__ x,
                             const float* __restrict__ dw,
                             const float* __restrict__ db,
                             float* __restrict__ out, int Kd, int N) {
  __shared__ float red[256];
  const int b = blockIdx.x;
  const int j = blockIdx.y;
  const _Float16* xb = x + (size_t)b * Kd;
  float s = 0.f;
  for (int i = threadIdx.x; i < Kd; i += blockDim.x)
    s += (float)xb[i] * dw[(size_t)i * N + j];
  red[threadIdx.x] = s;
  __syncthreads();
  for (int off = 128; off > 0; off >>= 1) {
    if ((int)threadIdx.x < off) red[threadIdx.x] += red[threadIdx.x + off];
    __syncthreads();
  }
  if (threadIdx.x == 0) out[b * N + j] = red[0] + db[j];
}

// ---------------------------------------------------------------------------
// Host-side orchestration
// ---------------------------------------------------------------------------
extern "C" void kernel_launch(void* const* d_in, const int* in_sizes, int n_in,
                              void* d_out, int out_size, void* d_ws,
                              size_t ws_size, hipStream_t stream) {
  (void)in_sizes; (void)n_in; (void)out_size; (void)ws_size;

  const float* imageA = (const float*)d_in[0];
  const float* imageB = (const float*)d_in[1];
  const float* w1  = (const float*)d_in[2];
  const float* b1  = (const float*)d_in[3];
  const float* w2  = (const float*)d_in[4];
  const float* b2  = (const float*)d_in[5];
  const float* w3  = (const float*)d_in[6];
  const float* b3  = (const float*)d_in[7];
  const float* cw1 = (const float*)d_in[8];
  const float* cb1 = (const float*)d_in[9];
  const float* g1  = (const float*)d_in[10];
  const float* be1 = (const float*)d_in[11];
  const float* m1  = (const float*)d_in[12];
  const float* v1  = (const float*)d_in[13];
  const float* cw2 = (const float*)d_in[14];
  const float* cb2 = (const float*)d_in[15];
  const float* g2  = (const float*)d_in[16];
  const float* be2 = (const float*)d_in[17];
  const float* m2  = (const float*)d_in[18];
  const float* v2  = (const float*)d_in[19];
  const float* dw  = (const float*)d_in[20];
  const float* db  = (const float*)d_in[21];
  float* outp = (float*)d_out;

  // Workspace layout (f16 slabs, 256B aligned).
  size_t off = 0;
  auto halloc = [&](size_t elems) {
    _Float16* p = (_Float16*)((char*)d_ws + off);
    off += ((elems * sizeof(_Float16) + 255) / 256) * 256;
    return p;
  };
  const int PS = 928;  // 900 correlation channels padded to a multiple of 32
  const int KW2  = 9 * 64;         // 576
  const int KW3  = 9 * 128;        // 1152
  const int KCW1 = 49 * PS;        // 45472
  const int KCW2 = 25 * 128;       // 3200
  _Float16* w2T    = halloc((size_t)128 * KW2);
  _Float16* w3T    = halloc((size_t)256 * KW3);
  _Float16* cw1T   = halloc((size_t)128 * KCW1);
  _Float16* cw2T   = halloc((size_t)64 * KCW2);
  _Float16* feat1h = halloc((size_t)16 * 120 * 120 * 64);   // reused A/B
  _Float16* feat2h = halloc((size_t)16 * 60 * 60 * 128);    // reused A/B
  _Float16* fAh    = halloc((size_t)16 * 900 * 256);
  _Float16* fBh    = halloc((size_t)16 * 900 * 256);
  _Float16* xh     = halloc((size_t)16 * 900 * PS);         // corr (padded)
  _Float16* y1h    = halloc((size_t)16 * 576 * 128);
  _Float16* y2h    = halloc((size_t)16 * 400 * 64);

  const int CV = 256;
  // Stage weights transposed [Cout, K] in f16 (with channel pad for cw1).
  stage_wT_kernel<<<(128 * KW2 + CV - 1) / CV, CV, 0, stream>>>(w2, w2T, 9, 64, 64, 128);
  stage_wT_kernel<<<(256 * KW3 + CV - 1) / CV, CV, 0, stream>>>(w3, w3T, 9, 128, 128, 256);
  stage_wT_kernel<<<(128 * KCW1 + CV - 1) / CV, CV, 0, stream>>>(cw1, cw1T, 49, 900, PS, 128);
  stage_wT_kernel<<<(64 * KCW2 + CV - 1) / CV, CV, 0, stream>>>(cw2, cw2T, 25, 128, 128, 64);

  // Feature extractor per image (intermediates reused).
  const float* imgs[2] = {imageA, imageB};
  _Float16*    fts[2]  = {fAh, fBh};
  for (int im = 0; im < 2; ++im) {
    conv1_direct_kernel<<<(16 * 120 * 120 * 64 + CV - 1) / CV, CV, 0, stream>>>(
        imgs[im], w1, b1, feat1h);
    // conv2: 120->60, 64->128, 3x3 s2, relu.  Mtiles=225 -> 57 blocks x4 waves
    conv_wmma_kernel<<<dim3(57, 2, 16), TB, 0, stream>>>(
        feat1h, w2T, b2, nullptr, nullptr, nullptr, nullptr, feat2h,
        120, 120, 64, 60, 60, 128, 3, 2, 0);
    // conv3: 60->30, 128->256, 3x3 s2, relu.  Mtiles=57 -> 15 blocks
    conv_wmma_kernel<<<dim3(15, 4, 16), TB, 0, stream>>>(
        feat2h, w3T, b3, nullptr, nullptr, nullptr, nullptr, fts[im],
        60, 60, 128, 30, 30, 256, 3, 2, 0);
    // channel L2-norm (len 256), in place.
    rownorm_kernel<<<16 * 900, 256, 0, stream>>>(fts[im], 256, 256, 0, 1e-6f);
  }

  // Correlation: per-batch 900x900 GEMM over 256 channels, padded row stride.
  corr_wmma_kernel<<<dim3(15, 15, 16), TB, 0, stream>>>(fAh, fBh, xh, 900, 256, PS);
  // Zero the 28 pad channels, then relu + L2-norm over the 900 real channels.
  pad_zero_kernel<<<(16 * 900 * (PS - 900) + CV - 1) / CV, CV, 0, stream>>>(
      xh, 16 * 900, 900, PS);
  rownorm_kernel<<<16 * 900, 256, 0, stream>>>(xh, 900, PS, 1, 1e-6f);

  // Regressor conv1: 30->24 VALID 7x7, Cin=928(padded), Cout=128, relu+BN.
  conv_wmma_kernel<<<dim3(9, 2, 16), TB, 0, stream>>>(
      xh, cw1T, cb1, g1, be1, m1, v1, y1h,
      30, 30, PS, 24, 24, 128, 7, 1, 1);
  // Regressor conv2: 24->20 VALID 5x5, Cin=128, Cout=64, relu+BN.
  conv_wmma_kernel<<<dim3(7, 1, 16), TB, 0, stream>>>(
      y1h, cw2T, cb2, g2, be2, m2, v2, y2h,
      24, 24, 128, 20, 20, 64, 5, 1, 1);

  // Dense head -> [16, 18] f32 (viewed as [16,9,2]).
  dense_kernel<<<dim3(16, 18), 256, 0, stream>>>(y2h, dw, db, outp, 25600, 18);
}
